// Gemma2Attention_52166672778080
// MI455X (gfx1250) — matrix operand order repaired and checked by reading, not compiled, not run on hardware
//
#include <hip/hip_runtime.h>
#include <hip/hip_bf16.h>

// ---------------- constants ----------------
#define S_LEN 2048
#define HID   3584
#define NH    16
#define NKV   8
#define HD    256
#define WIN   1024
#define CAPV  50.0f
#define SCL   0.0625f                 // 256^-0.5
#define LN_THETA 9.210340371976184f  // ln(10000)

typedef unsigned short u16;
typedef __attribute__((ext_vector_type(16))) __bf16 v16bf;
typedef __attribute__((ext_vector_type(8)))  float  v8f;

union Frag16 {
    uint4 u[2];
    v16bf v;
    u16   h[16];
};

static __device__ __forceinline__ u16 f2bf(float f) {
    unsigned int x = __float_as_uint(f);
    unsigned int r = x + 0x7FFFu + ((x >> 16) & 1u);   // RNE
    return (u16)(r >> 16);
}
static __device__ __forceinline__ float bf2f(u16 b) {
    return __uint_as_float(((unsigned int)b) << 16);
}
static __device__ __forceinline__ v8f wmma_bf16(v16bf a, v16bf b, v8f c) {
    return __builtin_amdgcn_wmma_f32_16x16x32_bf16(false, a, false, b, (short)0, c, false, false);
}

// ---------------- CDNA5 async global->LDS staging (guarded) ----------------
#if defined(__has_builtin)
# if __has_builtin(__builtin_amdgcn_global_load_async_to_lds_b128)
#  define HAVE_ASYNC 1
# endif
#endif
#ifndef HAVE_ASYNC
# define HAVE_ASYNC 0
#endif

typedef int vsi4 __attribute__((vector_size(16)));
typedef __attribute__((address_space(1))) vsi4 g_vsi4;
typedef __attribute__((address_space(3))) vsi4 l_vsi4;

static __device__ __forceinline__ void stage_b128(void* l, const void* g) {
#if HAVE_ASYNC
    __builtin_amdgcn_global_load_async_to_lds_b128(
        (g_vsi4*)((__attribute__((address_space(1))) void*)(void*)(g)),
        (l_vsi4*)((__attribute__((address_space(3))) void*)(l)), 0, 0);
#else
    *reinterpret_cast<uint4*>(l) = *reinterpret_cast<const uint4*>(g);
#endif
}

static __device__ __forceinline__ void wait_async() {
#if HAVE_ASYNC
# if __has_builtin(__builtin_amdgcn_s_wait_asynccnt)
    __builtin_amdgcn_s_wait_asynccnt(0);
# else
    asm volatile("s_wait_asynccnt 0" ::: "memory");
# endif
#endif
}

static __device__ __forceinline__ void wave_lds_fence() {
    // intra-wave LDS RAW: LDS is in-order per wave; just stop reordering + drain
    asm volatile("s_wait_dscnt 0" ::: "memory");
}

// ---------------- fp32 -> bf16 (plain) ----------------
__global__ void cvt_bf16_kernel(const float* __restrict__ in, u16* __restrict__ out, long long n) {
    long long i = (long long)blockIdx.x * blockDim.x + threadIdx.x;
    if (i < n) out[i] = f2bf(in[i]);
}

// ---------------- fp32 -> bf16 transposing: in[K][N] -> out[N][K] ----------------
__global__ __launch_bounds__(256) void cvt_t_kernel(const float* __restrict__ in,
                                                    u16* __restrict__ out, int K, int N) {
    __shared__ float tile[32][33];
    const int k0 = blockIdx.x * 32, n0 = blockIdx.y * 32;
    const int c = threadIdx.x & 31, r8 = threadIdx.x >> 5;
    #pragma unroll
    for (int i = 0; i < 4; ++i) {
        int r = r8 + i * 8;
        tile[r][c] = in[(size_t)(k0 + r) * N + n0 + c];
    }
    __syncthreads();
    #pragma unroll
    for (int i = 0; i < 4; ++i) {
        int r = r8 + i * 8;
        out[(size_t)(n0 + r) * K + k0 + c] = f2bf(tile[c][r]);
    }
}

// ---------------- bf16 transpose: in[R][C] -> out[C][R] ----------------
__global__ __launch_bounds__(256) void tr_bf16_kernel(const u16* __restrict__ in,
                                                      u16* __restrict__ out, int R, int C) {
    __shared__ u16 tile[32][34];
    const int r0 = blockIdx.x * 32, c0 = blockIdx.y * 32;
    const int c = threadIdx.x & 31, r8 = threadIdx.x >> 5;
    #pragma unroll
    for (int i = 0; i < 4; ++i) {
        int r = r8 + i * 8;
        tile[r][c] = in[(size_t)(r0 + r) * C + c0 + c];
    }
    __syncthreads();
    #pragma unroll
    for (int i = 0; i < 4; ++i) {
        int r = r8 + i * 8;
        out[(size_t)(c0 + r) * R + r0 + c] = tile[c][r];
    }
}

// ---------------- bf16 GEMM: C[M,N] = A[M,K] * BT[N,K]^T ----------------
// 256 thr = 8 waves; block tile 128x128; wave tile 32x64; K-chunk 32; double buffered.
template <bool OUT_BF16>
__global__ __launch_bounds__(256) void gemm_bf16(const u16* __restrict__ A,
                                                 const u16* __restrict__ BT,
                                                 void* __restrict__ Cp,
                                                 int M, int N, int K) {
    __shared__ __align__(16) u16 lA[2][128 * 40];
    __shared__ __align__(16) u16 lB[2][128 * 40];

    const int tid  = threadIdx.x;
    const int lane = tid & 31;
    const int wid  = tid >> 5;
    const int wm   = wid & 3;
    const int wn   = wid >> 2;
    const int lm   = lane & 15;
    const int lg   = lane >> 4;

    const int bm = blockIdx.y * 128;
    const int bn = blockIdx.x * 128;

    v8f acc[2][4];
    for (int a = 0; a < 2; ++a)
        for (int b = 0; b < 4; ++b)
            for (int r = 0; r < 8; ++r) acc[a][b][r] = 0.0f;

    const int rowL = tid >> 1, colL = (tid & 1) * 16;   // 128 rows x 32 cols, both tiles

    auto stage = [&](int k0, int s) {
        const u16* ga = A  + (size_t)(bm + rowL) * K + k0 + colL;
        const u16* gb = BT + (size_t)(bn + rowL) * K + k0 + colL;
        u16* da = &lA[s][rowL * 40 + colL];
        u16* db = &lB[s][rowL * 40 + colL];
        stage_b128(da,     ga);
        stage_b128(da + 8, ga + 8);
        stage_b128(db,     gb);
        stage_b128(db + 8, gb + 8);
    };

    const int nk = K >> 5;
    stage(0, 0);
    for (int i = 0; i < nk; ++i) {
        const int b = i & 1;
        wait_async();
        __syncthreads();
        if (i + 1 < nk) stage((i + 1) << 5, 1 - b);

        Frag16 af[2], bf[4];
        #pragma unroll
        for (int a = 0; a < 2; ++a) {
            const u16* base = &lA[b][(wm * 32 + a * 16 + lm) * 40];
            af[a].u[0] = *reinterpret_cast<const uint4*>(base + lg * 16);
            af[a].u[1] = *reinterpret_cast<const uint4*>(base + lg * 16 + 8);
        }
        #pragma unroll
        for (int bb = 0; bb < 4; ++bb) {
            const u16* base = &lB[b][(wn * 64 + bb * 16 + lm) * 40 + lg * 16];
            bf[bb].u[0] = *reinterpret_cast<const uint4*>(base);
            bf[bb].u[1] = *reinterpret_cast<const uint4*>(base + 8);
        }
        #pragma unroll
        for (int a = 0; a < 2; ++a)
            #pragma unroll
            for (int bb = 0; bb < 4; ++bb)
                acc[a][bb] = wmma_bf16(af[a].v, bf[bb].v, acc[a][bb]);
    }

    #pragma unroll
    for (int a = 0; a < 2; ++a)
        #pragma unroll
        for (int bb = 0; bb < 4; ++bb)
            #pragma unroll
            for (int r = 0; r < 8; ++r) {
                int row = bm + wm * 32 + a * 16 + r + 8 * lg;
                int col = bn + wn * 64 + bb * 16 + lm;
                if (OUT_BF16)
                    ((u16*)Cp)[(size_t)row * N + col] = f2bf(acc[a][bb][r]);
                else
                    ((float*)Cp)[(size_t)row * N + col] = acc[a][bb][r];
            }
}

// ---------------- in-place NeoX RoPE on bf16 [S, nh*256] ----------------
__global__ void rope_kernel(u16* __restrict__ x, const int* __restrict__ pos, int nh) {
    long long idx = (long long)blockIdx.x * blockDim.x + threadIdx.x;
    int d    = (int)(idx & 127);
    int head = (int)((idx >> 7) % nh);
    int s    = (int)(idx / (128LL * nh));
    if (s >= S_LEN) return;
    float inv = __expf((-2.0f * (float)d / (float)HD) * LN_THETA);
    float fr  = (float)pos[s] * inv;
    float sn, cs;
    __sincosf(fr, &sn, &cs);
    u16* base = x + (size_t)s * nh * HD + (size_t)head * HD;
    float x1 = bf2f(base[d]);
    float x2 = bf2f(base[d + 128]);
    base[d]       = f2bf(x1 * cs - x2 * sn);
    base[d + 128] = f2bf(x2 * cs + x1 * sn);
}

// ---------------- flash attention (sliding window + softcap, fixed max) ----------------
// grid (S/64, NH); block 128 = 4 waves; each wave owns 16 q-rows.
// vT layout: [NKV*HD][S]  (pre-transposed V)
__global__ __launch_bounds__(128) void attn_kernel(const u16* __restrict__ q,
                                                   const u16* __restrict__ k,
                                                   const u16* __restrict__ vT,
                                                   u16* __restrict__ attn) {
    __shared__ __align__(16) u16 lK[2][32 * 264];    // [k_row][d]
    __shared__ __align__(16) u16 lV[2][256 * 40];    // [d][k_row]
    __shared__ __align__(16) u16 lP[4 * 16 * 40];    // per-wave P tile [m][k]

    const int tid  = threadIdx.x;
    const int lane = tid & 31;
    const int wid  = tid >> 5;
    const int lm   = lane & 15;
    const int lg   = lane >> 4;

    const int qbase = blockIdx.x * 64;
    const int h     = blockIdx.y;
    const int kvh   = h >> 1;
    const int qrow0 = qbase + wid * 16;

    // Q fragments (d = 0..255, 8 chunks of 32) in registers
    Frag16 qf[8];
    const u16* qrow = q + (size_t)(qrow0 + lm) * (NH * HD) + (size_t)h * HD;
    #pragma unroll
    for (int dc = 0; dc < 8; ++dc) {
        qf[dc].u[0] = *reinterpret_cast<const uint4*>(qrow + dc * 32 + lg * 16);
        qf[dc].u[1] = *reinterpret_cast<const uint4*>(qrow + dc * 32 + lg * 16 + 8);
    }

    v8f acc[16];
    for (int nt = 0; nt < 16; ++nt)
        for (int r = 0; r < 8; ++r) acc[nt][r] = 0.0f;
    float lrow[8];
    #pragma unroll
    for (int r = 0; r < 8; ++r) lrow[r] = 0.0f;

    const int klo = (qbase - (WIN - 1)) > 0 ? (qbase - (WIN - 1)) : 0;
    const int kc0 = klo & ~31;
    const int khi = qbase + 63;
    const int nch = ((khi - kc0) >> 5) + 1;

    const int kr  = tid >> 2;            // 0..31
    const int kd0 = (tid & 3) * 64;      // K-tile d columns
    const int vd0 = tid * 2;             // V-tile d rows (2 per thread)

    auto stage = [&](int kc, int s) {
        const u16* ks = k + (size_t)(kc + kr) * (NKV * HD) + (size_t)kvh * HD + kd0;
        u16* kdst = &lK[s][kr * 264 + kd0];
        #pragma unroll
        for (int i = 0; i < 8; ++i) stage_b128(kdst + i * 8, ks + i * 8);
        #pragma unroll
        for (int j = 0; j < 2; ++j) {
            const u16* vs = vT + (size_t)(kvh * HD + vd0 + j) * S_LEN + kc;
            u16* vdst = &lV[s][(vd0 + j) * 40];
            #pragma unroll
            for (int i = 0; i < 4; ++i) stage_b128(vdst + i * 8, vs + i * 8);
        }
    };

    stage(kc0, 0);
    for (int i = 0; i < nch; ++i) {
        const int b  = i & 1;
        const int kc = kc0 + (i << 5);
        wait_async();
        __syncthreads();
        if (i + 1 < nch) stage(kc + 32, 1 - b);

        // scores (two 16x16 tiles), softcap, fixed-max exp, write P
        float pf[2][8];
        #pragma unroll
        for (int t = 0; t < 2; ++t) {
            v8f s;
            for (int r = 0; r < 8; ++r) s[r] = 0.0f;
            #pragma unroll
            for (int dc = 0; dc < 8; ++dc) {
                Frag16 bfr;
                const u16* base = &lK[b][(t * 16 + lm) * 264 + dc * 32 + lg * 16];
                bfr.u[0] = *reinterpret_cast<const uint4*>(base);
                bfr.u[1] = *reinterpret_cast<const uint4*>(base + 8);
                s = wmma_bf16(qf[dc].v, bfr.v, s);
            }
            #pragma unroll
            for (int r = 0; r < 8; ++r) {
                float x = s[r] * SCL;
                x = CAPV * tanhf(x * (1.0f / CAPV));      // x in [-CAP, CAP]
                int ki = kc + t * 16 + lm;
                int qi = qrow0 + r + 8 * lg;
                bool good = (ki <= qi) && ((qi - ki) < WIN);
                float p = good ? __expf(x - CAPV) : 0.0f; // fixed softmax max = CAP
                pf[t][r] = p;
                lrow[r] += p;                             // per-lane partial sum
            }
        }

        // P -> LDS (per-wave region), intra-wave fence, reload as A-fragment
        u16* pb = &lP[wid * 640];
        #pragma unroll
        for (int t = 0; t < 2; ++t)
            #pragma unroll
            for (int r = 0; r < 8; ++r)
                pb[(r + 8 * lg) * 40 + t * 16 + lm] = f2bf(pf[t][r]);
        wave_lds_fence();

        Frag16 pfr;
        {
            const u16* base = &lP[wid * 640 + lm * 40];
            pfr.u[0] = *reinterpret_cast<const uint4*>(base + lg * 16);
            pfr.u[1] = *reinterpret_cast<const uint4*>(base + lg * 16 + 8);
        }

        // PV: out[16 x 256] += P[16 x 32] * V[32 x 256]
        #pragma unroll
        for (int nt = 0; nt < 16; ++nt) {
            Frag16 bfr;
            const u16* base = &lV[b][(nt * 16 + lm) * 40 + lg * 16];
            bfr.u[0] = *reinterpret_cast<const uint4*>(base);
            bfr.u[1] = *reinterpret_cast<const uint4*>(base + 8);
            acc[nt] = wmma_bf16(pfr.v, bfr.v, acc[nt]);
        }
    }

    // single end-of-kernel row-sum butterfly (16-lane groups), normalize, store
    #pragma unroll
    for (int r = 0; r < 8; ++r) {
        float ps = lrow[r];
        ps += __shfl_xor(ps, 1, 32);
        ps += __shfl_xor(ps, 2, 32);
        ps += __shfl_xor(ps, 4, 32);
        ps += __shfl_xor(ps, 8, 32);
        float inv = 1.0f / ps;
        int row = qrow0 + r + 8 * lg;
        #pragma unroll
        for (int nt = 0; nt < 16; ++nt) {
            int col = h * HD + nt * 16 + lm;
            attn[(size_t)row * (NH * HD) + col] = f2bf(acc[nt][r] * inv);
        }
    }
}

// ---------------- host side ----------------
extern "C" void kernel_launch(void* const* d_in, const int* in_sizes, int n_in,
                              void* d_out, int out_size, void* d_ws, size_t ws_size,
                              hipStream_t stream) {
    const float* hidden = (const float*)d_in[0];
    const int*   pos    = (const int*)d_in[1];
    const float* Wq     = (const float*)d_in[2];
    const float* Wk     = (const float*)d_in[3];
    const float* Wv     = (const float*)d_in[4];
    const float* Wo     = (const float*)d_in[5];
    float* out = (float*)d_out;

    char* ws = (char*)d_ws;
    size_t off = 0;
    auto alloc = [&](size_t bytes) { char* p = ws + off; off += (bytes + 255) & ~(size_t)255; return p; };

    u16* hid_bf = (u16*)alloc((size_t)S_LEN * HID * 2);
    u16* wqT    = (u16*)alloc((size_t)HID * NH * HD * 2);     // [4096][3584]
    u16* wkT    = (u16*)alloc((size_t)HID * NKV * HD * 2);    // [2048][3584]
    u16* wvT    = (u16*)alloc((size_t)HID * NKV * HD * 2);
    u16* woT    = (u16*)alloc((size_t)NH * HD * HID * 2);     // [3584][4096]
    u16* q_bf   = (u16*)alloc((size_t)S_LEN * NH * HD * 2);
    u16* k_bf   = (u16*)alloc((size_t)S_LEN * NKV * HD * 2);
    u16* v_bf   = (u16*)alloc((size_t)S_LEN * NKV * HD * 2);
    u16* vT     = (u16*)alloc((size_t)S_LEN * NKV * HD * 2);  // [2048][2048] transposed
    u16* a_bf   = (u16*)alloc((size_t)S_LEN * NH * HD * 2);

    // conversions (weights transposed to [N][K] for GEMM staging)
    {
        long long n = (long long)S_LEN * HID;
        cvt_bf16_kernel<<<(int)((n + 255) / 256), 256, 0, stream>>>(hidden, hid_bf, n);
    }
    cvt_t_kernel<<<dim3(HID / 32, (NH * HD) / 32),  256, 0, stream>>>(Wq, wqT, HID, NH * HD);
    cvt_t_kernel<<<dim3(HID / 32, (NKV * HD) / 32), 256, 0, stream>>>(Wk, wkT, HID, NKV * HD);
    cvt_t_kernel<<<dim3(HID / 32, (NKV * HD) / 32), 256, 0, stream>>>(Wv, wvT, HID, NKV * HD);
    cvt_t_kernel<<<dim3((NH * HD) / 32, HID / 32),  256, 0, stream>>>(Wo, woT, NH * HD, HID);

    // Q/K/V projections -> bf16
    gemm_bf16<true><<<dim3((NH * HD) / 128, S_LEN / 128), 256, 0, stream>>>(
        hid_bf, wqT, q_bf, S_LEN, NH * HD, HID);
    gemm_bf16<true><<<dim3((NKV * HD) / 128, S_LEN / 128), 256, 0, stream>>>(
        hid_bf, wkT, k_bf, S_LEN, NKV * HD, HID);
    gemm_bf16<true><<<dim3((NKV * HD) / 128, S_LEN / 128), 256, 0, stream>>>(
        hid_bf, wvT, v_bf, S_LEN, NKV * HD, HID);

    // RoPE in place
    rope_kernel<<<(S_LEN * NH * 128) / 256, 256, 0, stream>>>(q_bf, pos, NH);
    rope_kernel<<<(S_LEN * NKV * 128) / 256, 256, 0, stream>>>(k_bf, pos, NKV);

    // V -> V^T  ([S][KV*HD] -> [KV*HD][S])
    tr_bf16_kernel<<<dim3(S_LEN / 32, (NKV * HD) / 32), 256, 0, stream>>>(
        v_bf, vT, S_LEN, NKV * HD);

    // attention
    attn_kernel<<<dim3(S_LEN / 64, NH), 128, 0, stream>>>(q_bf, k_bf, vT, a_bf);

    // output projection -> fp32 d_out
    gemm_bf16<false><<<dim3(HID / 128, S_LEN / 128), 256, 0, stream>>>(
        a_bf, woT, out, S_LEN, HID, NH * HD);
}
